// DragPositionNetMultiScale_81097572483734
// MI455X (gfx1250) — matrix operand order repaired and checked by reading, not compiled, not run on hardware
//
#include <hip/hip_runtime.h>

typedef __attribute__((ext_vector_type(16))) _Float16 v16h;
typedef __attribute__((ext_vector_type(8)))  _Float16 v8h;
typedef __attribute__((ext_vector_type(8)))  float    v8f;

#define BVV    32            // B*V
#define NPTS   1024
#define TOTALP (BVV * NPTS)  // 32768 points per side
#define POSDIM 32
#define IMGSZ  512

// ---------------------------------------------------------------------------
// fast silu: x * rcp(1 + exp(-x))  (v_exp_f32 + v_rcp_f32, no IEEE div chain)
__device__ __forceinline__ float silu_f(float x) {
  return x * __builtin_amdgcn_rcpf(1.0f + __expf(-x));
}

// reflect padding index (jnp.pad 'reflect': -1 -> 1, n -> n-2)
__device__ __forceinline__ int refl_idx(int i, int n) {
  if (i < 0) i = -i;
  if (i >= n) i = 2 * n - 2 - i;
  return i;
}

// ---------------------------------------------------------------------------
// Fourier embedding: [2][32768][32] f16.  Per point, per freq i:
// [sin(f*x0), sin(f*x1), cos(f*x0), cos(f*x1)]
__global__ void fourier_kernel(const float* __restrict__ dstart,
                               const float* __restrict__ dend,
                               _Float16* __restrict__ fe) {
  int idx = blockIdx.x * blockDim.x + threadIdx.x;
  if (idx >= 2 * TOTALP) return;
  int side = idx >> 15;
  int p    = idx & (TOTALP - 1);
  const float* src = side ? dend : dstart;
  float x0 = src[2 * p + 0];
  float x1 = src[2 * p + 1];
  _Float16* o = fe + (size_t)idx * POSDIM;
#pragma unroll
  for (int i = 0; i < 8; ++i) {
    float f  = __expf(0.575646273249f * (float)i);  // 100^(i/8)
    float a0 = f * x0, a1 = f * x1;
    o[i * 4 + 0] = (_Float16)__sinf(a0);
    o[i * 4 + 1] = (_Float16)__sinf(a1);
    o[i * 4 + 2] = (_Float16)__cosf(a0);
    o[i * 4 + 3] = (_Float16)__cosf(a1);
  }
}

// ---------------------------------------------------------------------------
// f32 [K,N] -> f16 transposed [N,K]  (so B fragments are contiguous in K)
__global__ void convert_transpose_kernel(const float* __restrict__ in,
                                         _Float16* __restrict__ out,
                                         int K, int N) {
  int i = blockIdx.x * blockDim.x + threadIdx.x;
  if (i >= K * N) return;
  int k = i / N;
  int n = i - k * N;
  out[(size_t)n * K + k] = (_Float16)in[i];
}

// ---------------------------------------------------------------------------
// zero fill
__global__ void fill_zero_kernel(float* __restrict__ p, long long n) {
  long long i = (long long)blockIdx.x * blockDim.x + threadIdx.x;
  if (i < n) p[i] = 0.0f;
}

// ---------------------------------------------------------------------------
// WMMA GEMM:  out = act(A[M,K] @ W[K,N] + bias), W given transposed WT[N,K].
// K is a compile-time constant so the K-loop fully unrolls: the scheduler can
// hoist load clauses across WMMAs (staggered loadcnt waits) and all address
// arithmetic becomes immediate offsets.
// Each wave computes a 16 x (16*NT) strip: one A fragment feeds NT WMMAs per
// 32-wide K step.  Fragment layouts used here (contiguous k, same order in A and B):
//   A (16x32 f16): lane row = lane&15, halves = concat(Ap[h*16..+7], Ap[h*16+8..+7])
//   B (32x16 f16): lane col = lane&15, K = k0 + 16*h + e  -> contiguous in WT
template <int SILU, int OUTF16, int NT, int K>
__global__ __launch_bounds__(256)
void gemm_wmma_kernel(const _Float16* __restrict__ A,
                      const _Float16* __restrict__ WT,   // [N,K] f16
                      const float* __restrict__ bias,
                      _Float16* __restrict__ outH,
                      float* __restrict__ outF,
                      int M, int N) {
  const int lane  = threadIdx.x & 31;
  const int wave  = threadIdx.x >> 5;
  const int strip = blockIdx.x * (blockDim.x >> 5) + wave;
  const int stripsN = N / (16 * NT);
  const int totalStrips = (M >> 4) * stripsN;
  if (strip >= totalStrips) return;
  const int tm  = strip / stripsN;
  const int tn0 = (strip - tm * stripsN) * NT;  // first 16-col tile of strip
  const int row = lane & 15;                    // A row / B,C column
  const int h   = lane >> 4;

  const _Float16* Arow = A + (size_t)(tm * 16 + row) * K + h * 16;

  // per-tile WT row base pointers (contiguous in K)
  const _Float16* Wrow[NT];
#pragma unroll
  for (int t = 0; t < NT; ++t) {
    Wrow[t] = WT + (size_t)((tn0 + t) * 16 + row) * K + 16 * h;
  }

  v8f acc[NT];
#pragma unroll
  for (int t = 0; t < NT; ++t) acc[t] = (v8f){};

#pragma unroll
  for (int k0 = 0; k0 < K; k0 += 32) {
    // ---- A fragment: two contiguous 16B chunks (immediate offsets)
    v8h alo = *(const v8h*)(Arow + k0);
    v8h ahi = *(const v8h*)(Arow + k0 + 8);
    v16h a;
#pragma unroll
    for (int e = 0; e < 8; ++e) { a[e] = alo[e]; a[8 + e] = ahi[e]; }

    // ---- issue ALL B fragment loads first (distinct registers)
    v16h bfrag[NT];
#pragma unroll
    for (int t = 0; t < NT; ++t) {
      bfrag[t] = *(const v16h*)(Wrow[t] + k0);
    }
    // ---- then run the WMMA sequence
#pragma unroll
    for (int t = 0; t < NT; ++t) {
      acc[t] = __builtin_amdgcn_wmma_f32_16x16x32_f16(
          /*neg_a=*/false, a, /*neg_b=*/false, bfrag[t],
          /*c_mod=*/(short)0, acc[t], /*reuse_a=*/false, /*reuse_b=*/false);
    }
  }

  // ---- epilogue: C VGPR r holds M = tm*16 + r + 8*h, N = col
#pragma unroll
  for (int t = 0; t < NT; ++t) {
    const int col = (tn0 + t) * 16 + row;
    const float bv = bias[col];
#pragma unroll
    for (int r = 0; r < 8; ++r) {
      int m   = tm * 16 + r + 8 * h;
      float x = acc[t][r] + bv;
      if (SILU) x = silu_f(x);
      if (OUTF16) outH[(size_t)m * N + col] = (_Float16)x;
      else        outF[(size_t)m * N + col] = x;
    }
  }
}

// ---------------------------------------------------------------------------
// Scatter-add: grid[bv, r, c, ch] += emb[p, ch]   (grid channel-last)
__global__ void scatter_kernel(const float* __restrict__ emb,
                               const float* __restrict__ coords,
                               float* __restrict__ grid, int C, int scale) {
  long long idx = (long long)blockIdx.x * blockDim.x + threadIdx.x;
  long long total = (long long)TOTALP * C;
  if (idx >= total) return;
  int p  = (int)(idx / C);
  int ch = (int)(idx - (long long)p * C);
  int bv = p >> 10;  // NPTS = 1024
  int dsz = IMGSZ / scale;
  int r = ((int)coords[2 * p + 0]) / dsz;
  int c = ((int)coords[2 * p + 1]) / dsz;
  r = min(max(r, 0), scale - 1);
  c = min(max(c, 0), scale - 1);
  size_t gi = ((((size_t)bv * scale + r) * scale + c) * C) + ch;
  unsafeAtomicAdd(&grid[gi], emb[(size_t)p * C + ch]);
}

// ---------------------------------------------------------------------------
// 5x5 depthwise gaussian blur (sigma=1), reflect pad, channel-last grid in,
// NCHW out.
__global__ void blur_kernel(const float* __restrict__ grid,
                            float* __restrict__ out, int C, int S) {
  long long idx = (long long)blockIdx.x * blockDim.x + threadIdx.x;
  long long total = (long long)BVV * C * S * S;
  if (idx >= total) return;
  int j  = (int)(idx % S);
  int i  = (int)((idx / S) % S);
  int ch = (int)((idx / ((long long)S * S)) % C);
  int bv = (int)(idx / ((long long)S * S * C));
  const float g5[5] = {0.05448868f, 0.24420136f, 0.40261996f,
                       0.24420136f, 0.05448868f};
  float acc = 0.0f;
#pragma unroll
  for (int di = 0; di < 5; ++di) {
    int ii = refl_idx(i + di - 2, S);
    float gr = g5[di];
#pragma unroll
    for (int dj = 0; dj < 5; ++dj) {
      int jj = refl_idx(j + dj - 2, S);
      acc += gr * g5[dj] * grid[(((size_t)bv * S + ii) * S + jj) * C + ch];
    }
  }
  out[idx] = acc;
}

// ---------------------------------------------------------------------------
extern "C" void kernel_launch(void* const* d_in, const int* in_sizes, int n_in,
                              void* d_out, int out_size, void* d_ws,
                              size_t ws_size, hipStream_t stream) {
  static const int SCALES[6] = {256, 128, 64, 32, 16, 8};
  static const int CH[6]     = {32, 32, 64, 128, 256, 512};  // C = channels//2

  const float* dstart = (const float*)d_in[0];
  const float* dend   = (const float*)d_in[1];

  // ---- workspace partition (bytes) ----
  char* w = (char*)d_ws;
  _Float16* fe  = (_Float16*)w;  w += (size_t)2 * TOTALP * POSDIM * 2;  // 4 MiB
  _Float16* w1t = (_Float16*)w;  w += (size_t)32 * 128 * 2;             // WT [128,32]
  _Float16* w2t = (_Float16*)w;  w += (size_t)128 * 256 * 2;            // WT [256,128]
  _Float16* w3t = (_Float16*)w;  w += (size_t)256 * 512 * 2;            // WT [C,256]
  _Float16* H1  = (_Float16*)w;  w += (size_t)TOTALP * 128 * 2;         // 8 MiB
  _Float16* H2  = (_Float16*)w;  w += (size_t)TOTALP * 256 * 2;         // 16 MiB
  float*    EMB = (float*)w;     w += (size_t)TOTALP * 512 * 4;         // 64 MiB
  float*    GRID = (float*)w;    // up to 256 MiB (scale 256, C 32)

  // output offsets: (outs_s..., outs_e...) flattened in return order
  long long soff[6];
  long long cum = 0;
  for (int i = 0; i < 6; ++i) {
    soff[i] = cum;
    cum += (long long)BVV * CH[i] * SCALES[i] * SCALES[i];
  }
  const long long S_TOTAL = cum;  // 99,614,720

  // ---- stage 1: fourier embeddings for both sides ----
  fourier_kernel<<<(2 * TOTALP + 255) / 256, 256, 0, stream>>>(dstart, dend, fe);

  // ---- per-scale pipeline ----
  for (int i = 0; i < 6; ++i) {
    const int C = CH[i], scale = SCALES[i];
    const float* W1 = (const float*)d_in[2 + 6 * i + 0];
    const float* b1 = (const float*)d_in[2 + 6 * i + 1];
    const float* W2 = (const float*)d_in[2 + 6 * i + 2];
    const float* b2 = (const float*)d_in[2 + 6 * i + 3];
    const float* W3 = (const float*)d_in[2 + 6 * i + 4];
    const float* b3 = (const float*)d_in[2 + 6 * i + 5];

    convert_transpose_kernel<<<(32 * 128 + 255) / 256, 256, 0, stream>>>(
        W1, w1t, 32, 128);
    convert_transpose_kernel<<<(128 * 256 + 255) / 256, 256, 0, stream>>>(
        W2, w2t, 128, 256);
    convert_transpose_kernel<<<(256 * C + 255) / 256, 256, 0, stream>>>(
        W3, w3t, 256, C);

    for (int side = 0; side < 2; ++side) {
      const _Float16* A0 = fe + (size_t)side * TOTALP * POSDIM;
      const float* coords = side ? dend : dstart;

      // GEMM 1: [32768,32] x [32,128] + b1, silu -> H1 (f16)
      {
        int strips = (TOTALP / 16) * (128 / 64);
        gemm_wmma_kernel<1, 1, 4, 32><<<strips / 8, 256, 0, stream>>>(
            A0, w1t, b1, H1, nullptr, TOTALP, 128);
      }
      // GEMM 2: [32768,128] x [128,256] + b2, silu -> H2 (f16)
      {
        int strips = (TOTALP / 16) * (256 / 64);
        gemm_wmma_kernel<1, 1, 4, 128><<<strips / 8, 256, 0, stream>>>(
            H1, w2t, b2, H2, nullptr, TOTALP, 256);
      }
      // GEMM 3: [32768,256] x [256,C] + b3 -> EMB (f32)
      if (C >= 64) {
        int strips = (TOTALP / 16) * (C / 64);
        gemm_wmma_kernel<0, 0, 4, 256><<<strips / 8, 256, 0, stream>>>(
            H2, w3t, b3, nullptr, EMB, TOTALP, C);
      } else {
        int strips = (TOTALP / 16) * (C / 32);
        gemm_wmma_kernel<0, 0, 2, 256><<<strips / 8, 256, 0, stream>>>(
            H2, w3t, b3, nullptr, EMB, TOTALP, C);
      }

      // zero grid, scatter, blur
      long long gn = (long long)BVV * scale * scale * C;
      fill_zero_kernel<<<(unsigned)((gn + 255) / 256), 256, 0, stream>>>(GRID, gn);

      long long sn = (long long)TOTALP * C;
      scatter_kernel<<<(unsigned)((sn + 255) / 256), 256, 0, stream>>>(
          EMB, coords, GRID, C, scale);

      float* outp = (float*)d_out + (side ? S_TOTAL : 0) + soff[i];
      blur_kernel<<<(unsigned)((gn + 255) / 256), 256, 0, stream>>>(
          GRID, outp, C, scale);
    }
  }
}
